// ALIGNN_14285061227027
// MI455X (gfx1250) — compile-verified, matrix operand order checked by reading, not run on hardware
//
#include <hip/hip_runtime.h>

// ---------------------------------------------------------------------------
// CDNA5 (gfx1250) ALIGNN forward pass.
//  - All [*,K]x[K,N] linears run through v_wmma_f32_16x16x32_bf16 (wave32).
//  - fp32 activations are converted to bf16 while staging into LDS (pairwise:
//    float2 global load -> packed 2xbf16 32-bit DS store); weights are
//    repacked fp32 -> bf16 transposed ([N][KP], KP = ceil64(K)) so every
//    lane's WMMA fragment is two aligned 16B LDS reads.
//  - Segment sums use global fp32 atomics; LN/SiLU are wave-per-row kernels.
//
// Input index map (assumption: top-level setup_inputs dict in insertion
// order; the nested "params" pytree flattened JAX-style = dict keys sorted
// alphabetically, lists in order):
//   0..111   alignn[4] x { edge:eggc(14), node:eggc(14) }
//            eggc leaf order: dg.b,dg.w,du.b,du.w,eg.b,eg.w,egn_b,egn_g,
//                             nb,ng,sg.b,sg.w,su.b,su.w
//   112..115 ang_mlp1 (b,be,g,w)     116..119 ang_mlp2
//   120..123 atom_mlp                124      atom_table [108,92]
//   125..128 edge_mlp1               129..132 edge_mlp2
//   133      fc.b   134 fc.w         135..190 gcn[4] x eggc(14)
//   191 atomic_number  192 r[E,3]  193 src  194 dst
//   195 y_mask_idx     196 lg_src  197 lg_dst
// ---------------------------------------------------------------------------

typedef __attribute__((ext_vector_type(16))) __bf16        v16bf;
typedef __attribute__((ext_vector_type(8)))  float         v8f;
typedef __attribute__((ext_vector_type(4)))  unsigned int  v4u;

union FragU { v4u q[2]; v16bf v; };

__device__ __forceinline__ unsigned short f2bf(float f) {
  unsigned int u = __float_as_uint(f);
  u += 0x7FFFu + ((u >> 16) & 1u);        // round-to-nearest-even
  return (unsigned short)(u >> 16);
}

__device__ __forceinline__ float wredsum(float x) {
#pragma unroll
  for (int off = 16; off > 0; off >>= 1) x += __shfl_xor(x, off, 32);
  return x;
}

__device__ __forceinline__ float siluf(float v) { return v / (1.f + __expf(-v)); }

#define LDSK 72   // padded bf16 row stride in LDS (144B: rows hit distinct banks)

// ------------------------------ WMMA GEMM ----------------------------------
// C[M,N] = A[M,K] (fp32) @ W[K,N] + bias, via bf16 WMMA, fp32 accumulate.
// WT is W transposed+padded to bf16: WT[n*KP + k], KP = ceil64(K).
// Requires K even (true for all layers here: 40/64/80/92/256).
template <int BN>
__global__ __launch_bounds__(256) void k_gemm(
    const float* __restrict__ A, int M, int K, int KP,
    const unsigned short* __restrict__ WT, const float* __restrict__ bias,
    float* __restrict__ C, int N) {
  alignas(16) __shared__ unsigned short lA[128 * LDSK];
  alignas(16) __shared__ unsigned short lB[BN * LDSK];
  const int tid  = threadIdx.x;
  const int wv   = tid >> 5, lane = tid & 31;
  const int m0   = blockIdx.x * 128;
  const int n0   = blockIdx.y * BN;
  const int m16  = lane & 15, hi = lane >> 4;

  v8f acc[BN / 16];
#pragma unroll
  for (int t = 0; t < BN / 16; ++t)
#pragma unroll
    for (int j = 0; j < 8; ++j) acc[t][j] = 0.f;

  for (int kc = 0; kc < KP; kc += 64) {
    // stage A tile (128 x 64), fp32 -> bf16, zero-padded.
    // pairwise: float2 load + one packed 32-bit DS store (K even).
#pragma unroll
    for (int it = 0; it < 16; ++it) {
      int e  = (it * 256 + tid) * 2;       // even element index in tile
      int rr = e >> 6, kk = e & 63;        // kk even
      int gm = m0 + rr, gk = kc + kk;
      float vx = 0.f, vy = 0.f;
      if (gm < M && gk < K) {
        float2 v2 = *(const float2*)(A + (size_t)gm * K + gk);
        vx = v2.x; vy = v2.y;
      }
      unsigned int pk = (unsigned int)f2bf(vx) | ((unsigned int)f2bf(vy) << 16);
      *(unsigned int*)(lA + rr * LDSK + kk) = pk;
    }
    // stage B tile (BN x 64) as 16B vector copies
#pragma unroll
    for (int it = 0; it < (BN * 8) / 256; ++it) {
      int e = it * 256 + tid;
      int rr = e >> 3, c = e & 7;
      *(v4u*)(lB + rr * LDSK + c * 8) =
          *(const v4u*)(WT + (size_t)(n0 + rr) * KP + kc + c * 8);
    }
    // prefetch next A chunk (global_prefetch_b8)
    if (kc + 64 < KP) {
      int rr = m0 + (tid >> 1);
      int kp = kc + 64 + (tid & 1) * 32;
      if (rr < M && kp < K) __builtin_prefetch(&A[(size_t)rr * K + kp], 0, 0);
    }
    __syncthreads();
#pragma unroll
    for (int kk = 0; kk < 64; kk += 32) {
      FragU fa;
      const unsigned short* pa = lA + (wv * 16 + m16) * LDSK + kk + hi * 8;
      fa.q[0] = *(const v4u*)pa;
      fa.q[1] = *(const v4u*)(pa + 16);
#pragma unroll
      for (int t = 0; t < BN / 16; ++t) {
        FragU fb;
        const unsigned short* pb = lB + (t * 16 + m16) * LDSK + kk + hi * 8;
        fb.q[0] = *(const v4u*)pb;
        fb.q[1] = *(const v4u*)(pb + 16);
        acc[t] = __builtin_amdgcn_wmma_f32_16x16x32_bf16(
            false, fa.v, false, fb.v, (short)0, acc[t], false, false);
      }
    }
    __syncthreads();
  }
  // C/D layout: VGPR j -> row j + 8*hi, col = lane&15 within 16x16 tile
#pragma unroll
  for (int t = 0; t < BN / 16; ++t) {
    int col = n0 + t * 16 + m16;
    float bv = bias ? bias[col] : 0.f;
#pragma unroll
    for (int j = 0; j < 8; ++j) {
      int row = m0 + wv * 16 + 8 * hi + j;
      if (row < M) C[(size_t)row * N + col] = acc[t][j] + bv;
    }
  }
}

// Repack W[K][N] fp32 -> WT[N][KP] bf16 (zero pad K..KP)
__global__ void k_pack_wt(const float* __restrict__ W, unsigned short* __restrict__ WT,
                          int K, int N, int KP) {
  int idx = blockIdx.x * 256 + threadIdx.x;
  if (idx >= N * KP) return;
  int n = idx / KP, k = idx % KP;
  WT[idx] = (k < K) ? f2bf(W[(size_t)k * N + n]) : (unsigned short)0;
}

// ------------------------- elementwise / graph kernels ---------------------
__global__ void k_zero(float* __restrict__ p, int n) {
  int i = blockIdx.x * 256 + threadIdx.x;
  if (i < n) p[i] = 0.f;
}

__global__ void k_atom_gather(const float* __restrict__ table, const int* __restrict__ an,
                              float* __restrict__ out, int n) {
  int idx = blockIdx.x * 256 + threadIdx.x;
  if (idx >= n * 92) return;
  int i = idx / 92, c = idx % 92;
  out[idx] = table[(size_t)an[i] * 92 + c];
}

__global__ void k_geom(const float* __restrict__ r, int E,
                       float* __restrict__ bl, float* __restrict__ fcut) {
  int e = blockIdx.x * 256 + threadIdx.x;
  if (e >= E) return;
  float x = r[3 * e], y = r[3 * e + 1], z = r[3 * e + 2];
  float l = sqrtf(x * x + y * y + z * z);
  bl[e] = l;
  const float on2 = 56.25f, cut2 = 64.f, inv = 1.f / 465.484375f;  // (cut2-on2)^3
  float r2 = l * l, t = cut2 - r2;
  float s = t * t * (cut2 + 2.f * r2 - 3.f * on2) * inv;
  fcut[e] = (l < 7.5f) ? 1.f : ((l < 8.f) ? s : 0.f);
}

__global__ void k_cos(const float* __restrict__ r, const int* __restrict__ ymask,
                      const int* __restrict__ lgs, const int* __restrict__ lgd,
                      float* __restrict__ cosv, int n_lg) {
  int t = blockIdx.x * 256 + threadIdx.x;
  if (t >= n_lg) return;
  int ea = ymask[lgs[t]], eb = ymask[lgd[t]];
  float x1 = -r[3 * ea], y1 = -r[3 * ea + 1], z1 = -r[3 * ea + 2];
  float x2 =  r[3 * eb], y2 =  r[3 * eb + 1], z2 =  r[3 * eb + 2];
  float n1 = sqrtf(x1 * x1 + y1 * y1 + z1 * z1);
  float n2 = sqrtf(x2 * x2 + y2 * y2 + z2 * z2);
  cosv[t] = (x1 * x2 + y1 * y2 + z1 * z2) / (n1 * n2);
}

__global__ void k_rbf(const float* __restrict__ d, float* __restrict__ out,
                      int n, int bins, float vmin, float vmax) {
  long long idx = (long long)blockIdx.x * 256 + threadIdx.x;
  if (idx >= (long long)n * bins) return;
  int i = (int)(idx / bins), b = (int)(idx % bins);
  float step = (vmax - vmin) / (bins - 1);
  float c = vmin + b * step;
  float gamma = 1.f / step;
  float df = d[i] - c;
  out[idx] = __expf(-gamma * df * df);
}

// X = silu(ln(X)*g + be), wave-per-row, W in {64,256}
__global__ __launch_bounds__(256) void k_ln_silu(float* __restrict__ X,
                                                 const float* __restrict__ g,
                                                 const float* __restrict__ be,
                                                 int M, int W) {
  int row = blockIdx.x * 8 + (threadIdx.x >> 5);
  int lane = threadIdx.x & 31;
  if (row >= M) return;
  float v[8];
  int nper = W >> 5;
  float s = 0.f;
  for (int i = 0; i < nper; ++i) { v[i] = X[(size_t)row * W + lane + 32 * i]; s += v[i]; }
  float mu = wredsum(s) / W;
  float q = 0.f;
  for (int i = 0; i < nper; ++i) { float dd = v[i] - mu; q += dd * dd; }
  float rs = rsqrtf(wredsum(q) / W + 1e-5f);
  for (int i = 0; i < nper; ++i) {
    int c = lane + 32 * i;
    X[(size_t)row * W + c] = siluf((v[i] - mu) * rs * g[c] + be[c]);
  }
}

// m = SG[src]+DG[dst]+EG (in place over EG); sigma=sigmoid(m)*fcut;
// NUM[dst]+=sigma*DU[src]; DEN[dst]+=sigma
__global__ void k_combine(const float* __restrict__ SG, const float* __restrict__ DG,
                          float* __restrict__ EGM, const float* __restrict__ DU,
                          const int* __restrict__ es, const int* __restrict__ ed,
                          const float* __restrict__ fcut,
                          float* __restrict__ NUM, float* __restrict__ DEN, int E) {
  long long idx = (long long)blockIdx.x * 256 + threadIdx.x;
  if (idx >= (long long)E * 256) return;
  int e = (int)(idx >> 8), c = (int)(idx & 255);
  int s = es[e], d = ed[e];
  float m = SG[(size_t)s * 256 + c] + DG[(size_t)d * 256 + c] + EGM[(size_t)e * 256 + c];
  EGM[(size_t)e * 256 + c] = m;
  float sig = 1.f / (1.f + __expf(-m));
  if (fcut) sig *= fcut[e];
  atomicAdd(&NUM[(size_t)d * 256 + c], sig * DU[(size_t)s * 256 + c]);
  atomicAdd(&DEN[(size_t)d * 256 + c], sig);
}

// Y += silu(skip ? m : ln(m)*g+b)
__global__ __launch_bounds__(256) void k_edge_out(float* __restrict__ Y,
                                                  const float* __restrict__ Mb,
                                                  const float* __restrict__ g,
                                                  const float* __restrict__ b,
                                                  int E, int skip) {
  int row = blockIdx.x * 8 + (threadIdx.x >> 5);
  int lane = threadIdx.x & 31;
  if (row >= E) return;
  float v[8];
  float s = 0.f;
  for (int i = 0; i < 8; ++i) { v[i] = Mb[(size_t)row * 256 + lane + 32 * i]; s += v[i]; }
  if (!skip) {
    float mu = wredsum(s) * (1.f / 256.f);
    float q = 0.f;
    for (int i = 0; i < 8; ++i) { float dd = v[i] - mu; q += dd * dd; }
    float rs = rsqrtf(wredsum(q) * (1.f / 256.f) + 1e-5f);
    for (int i = 0; i < 8; ++i) {
      int c = lane + 32 * i;
      v[i] = (v[i] - mu) * rs * g[c] + b[c];
    }
  }
  for (int i = 0; i < 8; ++i) {
    size_t o = (size_t)row * 256 + lane + 32 * i;
    Y[o] += siluf(v[i]);
  }
}

// X += silu(ln(SU + NUM/(DEN+1e-6))*ng + nb)
__global__ __launch_bounds__(256) void k_node_update(float* __restrict__ X,
                                                     const float* __restrict__ SU,
                                                     const float* __restrict__ NUM,
                                                     const float* __restrict__ DEN,
                                                     const float* __restrict__ ng,
                                                     const float* __restrict__ nb,
                                                     int M) {
  int row = blockIdx.x * 8 + (threadIdx.x >> 5);
  int lane = threadIdx.x & 31;
  if (row >= M) return;
  float v[8];
  float s = 0.f;
  for (int i = 0; i < 8; ++i) {
    size_t o = (size_t)row * 256 + lane + 32 * i;
    v[i] = SU[o] + NUM[o] / (DEN[o] + 1e-6f);
    s += v[i];
  }
  float mu = wredsum(s) * (1.f / 256.f);
  float q = 0.f;
  for (int i = 0; i < 8; ++i) { float dd = v[i] - mu; q += dd * dd; }
  float rs = rsqrtf(wredsum(q) * (1.f / 256.f) + 1e-5f);
  for (int i = 0; i < 8; ++i) {
    int c = lane + 32 * i;
    size_t o = (size_t)row * 256 + c;
    X[o] += siluf((v[i] - mu) * rs * ng[c] + nb[c]);
  }
}

__global__ void k_gather_rows(float* __restrict__ out, const float* __restrict__ in,
                              const int* __restrict__ idx, int n) {
  long long t = (long long)blockIdx.x * 256 + threadIdx.x;
  if (t >= (long long)n * 256) return;
  int i = (int)(t >> 8), c = (int)(t & 255);
  out[(size_t)i * 256 + c] = in[(size_t)idx[i] * 256 + c];
}

__global__ void k_scatter_rows(float* __restrict__ out, const float* __restrict__ in,
                               const int* __restrict__ idx, int n) {
  long long t = (long long)blockIdx.x * 256 + threadIdx.x;
  if (t >= (long long)n * 256) return;
  int i = (int)(t >> 8), c = (int)(t & 255);
  out[(size_t)idx[i] * 256 + c] = in[(size_t)i * 256 + c];
}

__global__ void k_readout(const float* __restrict__ X, const float* __restrict__ w,
                          const float* __restrict__ b, int n, float* __restrict__ out) {
  __shared__ float sred[256];
  float acc = 0.f;
  for (int i = threadIdx.x; i < n; i += 256) {
    const float* xr = X + (size_t)i * 256;
    float d = 0.f;
    for (int c = 0; c < 256; ++c) d += xr[c] * w[c];
    acc += d;
  }
  sred[threadIdx.x] = acc;
  __syncthreads();
  for (int s = 128; s > 0; s >>= 1) {
    if (threadIdx.x < s) sred[threadIdx.x] += sred[threadIdx.x + s];
    __syncthreads();
  }
  if (threadIdx.x == 0) out[0] = sred[0] / (float)n + b[0];
}

// ------------------------------- host side ---------------------------------
namespace {
struct Lin  { const float *b, *w; };
struct Mlp  { const float *b, *be, *g, *w; };
struct Eggc { Lin dg, du, eg; const float *egn_b, *egn_g, *nb, *ng; Lin sg, su; };

inline const float* F(void* const* d, int i) { return (const float*)d[i]; }

inline Eggc eggc_at(void* const* d, int base) {
  Eggc p;
  p.dg = {F(d, base + 0), F(d, base + 1)};
  p.du = {F(d, base + 2), F(d, base + 3)};
  p.eg = {F(d, base + 4), F(d, base + 5)};
  p.egn_b = F(d, base + 6); p.egn_g = F(d, base + 7);
  p.nb = F(d, base + 8);    p.ng = F(d, base + 9);
  p.sg = {F(d, base + 10), F(d, base + 11)};
  p.su = {F(d, base + 12), F(d, base + 13)};
  return p;
}
inline Mlp mlp_at(void* const* d, int base) {
  return Mlp{F(d, base), F(d, base + 1), F(d, base + 2), F(d, base + 3)};
}
inline unsigned cdiv(long long a, long long b) { return (unsigned)((a + b - 1) / b); }
}  // namespace

extern "C" void kernel_launch(void* const* d_in, const int* in_sizes, int n_in,
                              void* d_out, int out_size, void* d_ws, size_t ws_size,
                              hipStream_t stream) {
  (void)n_in; (void)out_size; (void)ws_size;

  const int   n_nodes = in_sizes[191];
  const int*  an      = (const int*)d_in[191];
  const float* r      = (const float*)d_in[192];
  const int*  srcI    = (const int*)d_in[193];
  const int   E       = in_sizes[193];
  const int*  dstI    = (const int*)d_in[194];
  const int*  ymask   = (const int*)d_in[195];
  const int   n_sub   = in_sizes[195];
  const int*  lg_src  = (const int*)d_in[196];
  const int   n_lg    = in_sizes[196];
  const int*  lg_dst  = (const int*)d_in[197];

  const Mlp   ang1 = mlp_at(d_in, 112), ang2 = mlp_at(d_in, 116);
  const Mlp   amlp = mlp_at(d_in, 120);
  const float* atab = F(d_in, 124);
  const Mlp   em1 = mlp_at(d_in, 125), em2 = mlp_at(d_in, 129);
  const float* fc_b = F(d_in, 133);
  const float* fc_w = F(d_in, 134);

  const int maxA = (n_nodes > n_sub) ? n_nodes : n_sub;
  const int maxB = (E > n_lg) ? E : n_lg;

  // -------- workspace carve (256B aligned) --------
  char* wsp = (char*)d_ws;
  auto alloc = [&](size_t bytes) -> void* {
    void* p = wsp;
    wsp += (bytes + 255) & ~(size_t)255;
    return p;
  };
  float* X    = (float*)alloc((size_t)n_nodes * 256 * 4);
  float* Y    = (float*)alloc((size_t)E * 256 * 4);
  float* Z    = (float*)alloc((size_t)n_lg * 256 * 4);
  float* T_SG = (float*)alloc((size_t)maxA * 256 * 4);
  float* T_DG = (float*)alloc((size_t)maxA * 256 * 4);
  float* T_SU = (float*)alloc((size_t)maxA * 256 * 4);
  float* T_DU = (float*)alloc((size_t)maxA * 256 * 4);
  float* T_EGM = (float*)alloc((size_t)maxB * 256 * 4);   // EG, then m in place
  float* NUM  = (float*)alloc((size_t)maxA * 256 * 4);
  float* DEN  = (float*)alloc((size_t)maxA * 256 * 4);
  float* XSUB = (float*)alloc((size_t)n_sub * 256 * 4);
  float* TMP64 = (float*)alloc((size_t)maxB * 64 * 4);    // MLP hidden / atom-92
  float* BL   = (float*)alloc((size_t)E * 4);
  float* FCUT = (float*)alloc((size_t)E * 4);
  float* COS  = (float*)alloc((size_t)n_lg * 4);
  unsigned short* WT = (unsigned short*)alloc(256 * 256 * 2);

  auto gemm = [&](const float* A, int M, int K, const float* W, const float* bias,
                  float* C, int N) {
    int KP = (K + 63) & ~63;
    k_pack_wt<<<cdiv((long long)N * KP, 256), 256, 0, stream>>>(W, WT, K, N, KP);
    if ((N & 127) == 0)
      k_gemm<128><<<dim3(cdiv(M, 128), N / 128), 256, 0, stream>>>(A, M, K, KP, WT, bias, C, N);
    else
      k_gemm<64><<<dim3(cdiv(M, 128), N / 64), 256, 0, stream>>>(A, M, K, KP, WT, bias, C, N);
  };

  auto run_eggc = [&](const Eggc& p, const int* es, const int* ed, float* Xb, int nx,
                      float* Yb, int ne, const float* fcut, bool skip) {
    gemm(Xb, nx, 256, p.sg.w, p.sg.b, T_SG, 256);
    gemm(Xb, nx, 256, p.dg.w, p.dg.b, T_DG, 256);
    gemm(Xb, nx, 256, p.su.w, p.su.b, T_SU, 256);
    gemm(Xb, nx, 256, p.du.w, p.du.b, T_DU, 256);
    gemm(Yb, ne, 256, p.eg.w, p.eg.b, T_EGM, 256);
    k_zero<<<cdiv((long long)nx * 256, 256), 256, 0, stream>>>(NUM, nx * 256);
    k_zero<<<cdiv((long long)nx * 256, 256), 256, 0, stream>>>(DEN, nx * 256);
    k_combine<<<cdiv((long long)ne * 256, 256), 256, 0, stream>>>(
        T_SG, T_DG, T_EGM, T_DU, es, ed, fcut, NUM, DEN, ne);
    k_edge_out<<<cdiv(ne, 8), 256, 0, stream>>>(Yb, T_EGM, p.egn_g, p.egn_b, ne, skip ? 1 : 0);
    k_node_update<<<cdiv(nx, 8), 256, 0, stream>>>(Xb, T_SU, NUM, DEN, p.ng, p.nb, nx);
  };

  // -------- atom features: x = mlp(atom_table[an]) --------
  k_atom_gather<<<cdiv((long long)n_nodes * 92, 256), 256, 0, stream>>>(atab, an, TMP64, n_nodes);
  gemm(TMP64, n_nodes, 92, amlp.w, amlp.b, X, 256);
  k_ln_silu<<<cdiv(n_nodes, 8), 256, 0, stream>>>(X, amlp.g, amlp.be, n_nodes, 256);

  // -------- edge features: y = mlp2(mlp1(rbf(bl))) --------
  k_geom<<<cdiv(E, 256), 256, 0, stream>>>(r, E, BL, FCUT);
  k_rbf<<<cdiv((long long)E * 80, 256), 256, 0, stream>>>(BL, T_EGM, E, 80, 0.f, 8.f);
  gemm(T_EGM, E, 80, em1.w, em1.b, TMP64, 64);
  k_ln_silu<<<cdiv(E, 8), 256, 0, stream>>>(TMP64, em1.g, em1.be, E, 64);
  gemm(TMP64, E, 64, em2.w, em2.b, Y, 256);
  k_ln_silu<<<cdiv(E, 8), 256, 0, stream>>>(Y, em2.g, em2.be, E, 256);

  // -------- angle features: z = mlp2(mlp1(rbf(cos))) --------
  k_cos<<<cdiv(n_lg, 256), 256, 0, stream>>>(r, ymask, lg_src, lg_dst, COS, n_lg);
  k_rbf<<<cdiv((long long)n_lg * 40, 256), 256, 0, stream>>>(COS, T_EGM, n_lg, 40, -1.f, 1.f);
  gemm(T_EGM, n_lg, 40, ang1.w, ang1.b, TMP64, 64);
  k_ln_silu<<<cdiv(n_lg, 8), 256, 0, stream>>>(TMP64, ang1.g, ang1.be, n_lg, 64);
  gemm(TMP64, n_lg, 64, ang2.w, ang2.b, Z, 256);
  k_ln_silu<<<cdiv(n_lg, 8), 256, 0, stream>>>(Z, ang2.g, ang2.be, n_lg, 256);

  // -------- 4 ALIGNN layers (node-EGGC then line-graph edge-EGGC) --------
  for (int i = 0; i < 4; ++i) {
    Eggc pe = eggc_at(d_in, i * 28);        // alignn[i].edge
    Eggc pn = eggc_at(d_in, i * 28 + 14);   // alignn[i].node
    run_eggc(pn, srcI, dstI, X, n_nodes, Y, E, FCUT, i == 3);
    k_gather_rows<<<cdiv((long long)n_sub * 256, 256), 256, 0, stream>>>(XSUB, Y, ymask, n_sub);
    run_eggc(pe, lg_src, lg_dst, XSUB, n_sub, Z, n_lg, nullptr, false);
    k_scatter_rows<<<cdiv((long long)n_sub * 256, 256), 256, 0, stream>>>(Y, XSUB, ymask, n_sub);
  }
  // -------- 4 GCN layers --------
  for (int i = 0; i < 4; ++i) {
    Eggc gp = eggc_at(d_in, 135 + i * 14);
    run_eggc(gp, srcI, dstI, X, n_nodes, Y, E, FCUT, i == 3);
  }

  // -------- readout: mean(x @ fc.w) + fc.b --------
  k_readout<<<1, 256, 0, stream>>>(X, fc_w, fc_b, n_nodes, (float*)d_out);
}
